// FlashPhiAttention_57002805952805
// MI455X (gfx1250) — matrix operand order repaired and checked by reading, not compiled, not run on hardware
//
#include <hip/hip_runtime.h>
#include <hip/hip_bf16.h>

#define H_   32
#define KV_  8
#define HD_  128
#define D_   4096
#define B_   4
#define S_   1024
#define T_   (B_*S_)
#define QKVN ((H_ + 2*KV_) * HD_)   /* 6144 */
#define GQA  (H_/KV_)               /* 4 */
#define SCALE_ 0.08838834764831845f /* 128^-0.5 */

typedef __bf16 bf16;
typedef __attribute__((ext_vector_type(16))) __bf16 v16bf;
typedef __attribute__((ext_vector_type(8)))  float  v8f;

__device__ __forceinline__ v8f wmma_bf16(v16bf a, v16bf b, v8f c) {
  return __builtin_amdgcn_wmma_f32_16x16x32_bf16(
      /*neg_a=*/false, a, /*neg_b=*/false, b,
      /*c_mod=*/(short)0, c, /*reuse_a=*/false, /*reuse_b=*/false);
}

union FragU { v16bf v; uint4 q[2]; };

// A fragment (16x32, MxK): lane row = m, element e<8 -> K = e + 8*half,
// e>=8 -> K = 16 + (e-8) + 8*half. Two contiguous 16B loads per lane.
__device__ __forceinline__ v16bf frag_a(const bf16* row, int half) {
  FragU u;
  u.q[0] = *(const uint4*)(row + 8*half);
  u.q[1] = *(const uint4*)(row + 16 + 8*half);
  return u.v;
}
// B fragment (32x16, KxN) from LDS stored as B^T[n][k]:
// lane col n = lane%16, element e<8 -> K = e + 8*half,
// e>=8 -> K = 16 + (e-8) + 8*half. Two contiguous 16B loads per lane.
__device__ __forceinline__ v16bf frag_b(const bf16* row, int half) {
  FragU u;
  u.q[0] = *(const uint4*)(row + 8*half);
  u.q[1] = *(const uint4*)(row + 16 + 8*half);
  return u.v;
}

// Flat LDS address carries the LDS byte offset in addr[31:0] (ISA 10.2).
__device__ __forceinline__ unsigned lds_off(const void* p) {
  return (unsigned)(uintptr_t)p;
}
// CDNA5 async DMA: global -> LDS, 16B per lane, tracked with ASYNCcnt.
__device__ __forceinline__ void async_g2lds_b128(unsigned ldsoff, const void* gaddr) {
  asm volatile("global_load_async_to_lds_b128 %0, %1, off"
               :: "v"(ldsoff), "v"(gaddr) : "memory");
}
#define WAIT_ASYNC0() asm volatile("s_wait_asynccnt 0x0" ::: "memory")
#define WAIT_ASYNC4() asm volatile("s_wait_asynccnt 0x4" ::: "memory")

// ---------------- f32 -> bf16 convert ----------------
__global__ void cvt_f32_bf16(const float* __restrict__ in, bf16* __restrict__ out, int n4) {
  int i = blockIdx.x * blockDim.x + threadIdx.x;
  if (i < n4) {
    float4 f = ((const float4*)in)[i];
    bf16 o[4] = { (bf16)f.x, (bf16)f.y, (bf16)f.z, (bf16)f.w };
    *(uint2*)(out + 4*i) = *(const uint2*)o;
  }
}

// ---------------- generic bf16 GEMM: C = A(MxK) @ W(NxK)^T + bias ----------------
// Double-buffered LDS tiles streamed with async global->LDS DMA so the next
// K-tile copy overlaps the current tile's WMMA chain.
template<bool OUT_BF16>
__global__ __launch_bounds__(256)
void gemm_bf16(const bf16* __restrict__ A, const bf16* __restrict__ W,
               const float* __restrict__ bias, void* __restrict__ Cout,
               int M, int N, int K) {
  __shared__ __align__(16) bf16 As[2][128*32];
  __shared__ __align__(16) bf16 Bs[2][128*32];

  const int tid  = threadIdx.x;
  const int lane = tid & 31;
  const int wv   = tid >> 5;
  const int half = lane >> 4;
  const int l16  = lane & 15;
  const int wm   = wv & 3;        // 4 waves along M (32 rows each)
  const int wn   = wv >> 2;       // 2 waves along N (64 cols each)
  const int m0   = blockIdx.y * 128;
  const int n0   = blockIdx.x * 128;

  v8f acc[2][4];
  #pragma unroll
  for (int i = 0; i < 2; ++i)
    #pragma unroll
    for (int j = 0; j < 4; ++j) acc[i][j] = (v8f){};

  // per-thread staging coordinates: 512 uint4 per tile -> 2 per thread per array
  const int nk = K >> 5;
  int buf = 0;
  // prologue: stage tile 0 (4 async loads per thread)
  #pragma unroll
  for (int t = 0; t < 2; ++t) {
    int idx = tid + t*256;
    int r = idx >> 2, c = idx & 3;
    async_g2lds_b128(lds_off(&As[0][idx*8]), A + (size_t)(m0 + r)*K + c*8);
    async_g2lds_b128(lds_off(&Bs[0][idx*8]), W + (size_t)(n0 + r)*K + c*8);
  }

  for (int kt = 0; kt < nk; ++kt) {
    if (kt + 1 < nk) {
      const int k1 = (kt + 1) * 32;
      #pragma unroll
      for (int t = 0; t < 2; ++t) {
        int idx = tid + t*256;
        int r = idx >> 2, c = idx & 3;
        async_g2lds_b128(lds_off(&As[buf^1][idx*8]), A + (size_t)(m0 + r)*K + k1 + c*8);
        async_g2lds_b128(lds_off(&Bs[buf^1][idx*8]), W + (size_t)(n0 + r)*K + k1 + c*8);
      }
      WAIT_ASYNC4();    // current tile's 4 loads done; next tile's 4 still in flight
    } else {
      WAIT_ASYNC0();
    }
    __syncthreads();

    v16bf af[2], bw[4];
    #pragma unroll
    for (int i = 0; i < 2; ++i)
      af[i] = frag_a(&As[buf][(wm*32 + i*16 + l16)*32], half);
    #pragma unroll
    for (int j = 0; j < 4; ++j)
      bw[j] = frag_b(&Bs[buf][(wn*64 + j*16 + l16)*32], half);
    #pragma unroll
    for (int i = 0; i < 2; ++i)
      #pragma unroll
      for (int j = 0; j < 4; ++j)
        acc[i][j] = wmma_bf16(af[i], bw[j], acc[i][j]);

    __syncthreads();    // all waves done with buf before it is re-staged
    buf ^= 1;
  }

  #pragma unroll
  for (int i = 0; i < 2; ++i)
    #pragma unroll
    for (int j = 0; j < 4; ++j) {
      int n = n0 + wn*64 + j*16 + l16;
      float bv = bias[n];
      #pragma unroll
      for (int r = 0; r < 8; ++r) {
        int m = m0 + wm*32 + i*16 + r + 8*half;
        float v = acc[i][j][r] + bv;
        if (OUT_BF16) ((bf16*)Cout)[(size_t)m*N + n] = (bf16)v;
        else          ((float*)Cout)[(size_t)m*N + n] = v;
      }
    }
}

// ---------------- in-place RoPE on q/k columns of raw qkv ----------------
__global__ void rope_kernel(bf16* __restrict__ qkv,
                            const float* __restrict__ cosb,
                            const float* __restrict__ sinb) {
  int id = blockIdx.x * blockDim.x + threadIdx.x;
  if (id >= T_ * 40 * 32) return;       // 40 = H + KV heads, 32 = RD/2 pairs
  int d  = id & 31;
  int hh = (id >> 5) % 40;              // heads 0..31 = q, 32..39 = k
  int t  = id / (32 * 40);
  size_t base = (size_t)t * QKVN + hh * HD_;
  float x1 = (float)qkv[base + d];
  float x2 = (float)qkv[base + d + 32];
  float c = cosb[t*32 + d], s = sinb[t*32 + d];
  qkv[base + d]      = (bf16)(x1*c - x2*s);
  qkv[base + d + 32] = (bf16)(x2*c + x1*s);
}

// ---------------- causal GQA flash attention ----------------
__global__ __launch_bounds__(256)
void flash_kernel(const bf16* __restrict__ qkv, bf16* __restrict__ attn) {
  __shared__ __align__(16) bf16 Qs[128*128];   // 32 KB
  __shared__ __align__(16) bf16 Ks[32*128];    // 8 KB  [key][d]
  __shared__ __align__(16) bf16 Vt[128*32];    // 8 KB  [d][key]
  __shared__ __align__(16) bf16 Ps[8][16*32];  // 8 KB  per-wave probs

  const int tid  = threadIdx.x;
  const int lane = tid & 31;
  const int wv   = tid >> 5;
  const int half = lane >> 4;
  const int l16  = lane & 15;
  const int q0   = blockIdx.x * 128;
  const int b    = blockIdx.y;
  const int head = blockIdx.z;
  const int kvh  = head / GQA;

  const bf16* qg = qkv + (size_t)(b*S_)*QKVN + head*HD_;
  const bf16* kg = qkv + (size_t)(b*S_)*QKVN + H_*HD_ + kvh*HD_;
  const bf16* vg = qkv + (size_t)(b*S_)*QKVN + (H_+KV_)*HD_ + kvh*HD_;

  // stage 128x128 Q tile via async DMA (completion covered by first loop wait)
  #pragma unroll
  for (int t = 0; t < 8; ++t) {
    int idx = tid + t*256;              // 2048 x 16B
    int r = idx >> 4, c = idx & 15;
    async_g2lds_b128(lds_off(&Qs[r*128 + c*8]), qg + (size_t)(q0 + r)*QKVN + c*8);
  }

  float m_run[8], l_run[8];
  v8f oacc[8];
  #pragma unroll
  for (int r = 0; r < 8; ++r) { m_run[r] = -1e30f; l_run[r] = 0.0f; }
  #pragma unroll
  for (int j = 0; j < 8; ++j) oacc[j] = (v8f){};

  const int wave_qmax = q0 + wv*16 + 15;
  const int nblocks   = (q0 + 128) >> 5;

  for (int it = 0; it < nblocks; ++it) {
    const int kbs = it * 32;
    __syncthreads();
    // stage K block (32x128) via async DMA
    #pragma unroll
    for (int t = 0; t < 2; ++t) {
      int idx = tid + t*256;            // 512 x 16B
      int r = idx >> 4, c = idx & 15;
      async_g2lds_b128(lds_off(&Ks[r*128 + c*8]), kg + (size_t)(kbs + r)*QKVN + c*8);
    }
    // stage V block transposed into Vt[d][key] (VGPR path: transpose on the way)
    #pragma unroll
    for (int t = 0; t < 2; ++t) {
      int idx = tid + t*256;            // 512 groups of 8 elems along d
      int key = idx >> 4, d8 = (idx & 15) * 8;
      union { uint4 q; bf16 e[8]; } u;
      u.q = *(const uint4*)(vg + (size_t)(kbs + key)*QKVN + d8);
      #pragma unroll
      for (int e = 0; e < 8; ++e) Vt[(d8 + e)*32 + key] = u.e[e];
    }
    WAIT_ASYNC0();                      // Q (first iter) + K complete
    __syncthreads();

    const bool active = (kbs <= wave_qmax);   // wave-uniform -> EXEC all-ones
    if (active) {
      // scores: 16 queries x 32 keys, contraction over HD=128
      v8f s0 = (v8f){}, s1 = (v8f){};
      #pragma unroll
      for (int kk = 0; kk < 4; ++kk) {
        v16bf aq  = frag_a(&Qs[(wv*16 + l16)*128 + kk*32], half);
        v16bf kb0 = frag_b(&Ks[(l16)*128 + kk*32], half);
        v16bf kb1 = frag_b(&Ks[(16 + l16)*128 + kk*32], half);
        s0 = wmma_bf16(aq, kb0, s0);
        s1 = wmma_bf16(aq, kb1, s1);
      }
      // online softmax, row-wise (rows live in 16-lane halves)
      #pragma unroll
      for (int r = 0; r < 8; ++r) {
        int mrow = q0 + wv*16 + r + 8*half;
        int c0 = kbs + l16, c1 = kbs + 16 + l16;
        float sc0 = (c0 <= mrow) ? s0[r]*SCALE_ : -1e30f;
        float sc1 = (c1 <= mrow) ? s1[r]*SCALE_ : -1e30f;
        float mx = fmaxf(sc0, sc1);
        #pragma unroll
        for (int msk = 1; msk < 16; msk <<= 1)
          mx = fmaxf(mx, __shfl_xor(mx, msk, 32));
        float mnew = fmaxf(m_run[r], mx);
        float corr = __expf(m_run[r] - mnew);
        float p0 = __expf(sc0 - mnew);
        float p1 = __expf(sc1 - mnew);
        float ps = p0 + p1;
        #pragma unroll
        for (int msk = 1; msk < 16; msk <<= 1)
          ps += __shfl_xor(ps, msk, 32);
        l_run[r] = l_run[r]*corr + ps;
        m_run[r] = mnew;
        #pragma unroll
        for (int j = 0; j < 8; ++j) oacc[j][r] *= corr;
        int mloc = r + 8*half;
        Ps[wv][mloc*32 + l16]      = (bf16)p0;
        Ps[wv][mloc*32 + 16 + l16] = (bf16)p1;
      }
    }
    __syncthreads();   // make cross-lane Ps writes visible before A-frag reads
    if (active) {
      v16bf ap = frag_a(&Ps[wv][l16*32], half);
      #pragma unroll
      for (int j = 0; j < 8; ++j) {
        v16bf bv = frag_b(&Vt[(j*16 + l16)*32], half);
        oacc[j] = wmma_bf16(ap, bv, oacc[j]);
      }
    }
  }

  // epilogue: normalize and write attn (T x H*HD, head-major columns)
  #pragma unroll
  for (int r = 0; r < 8; ++r) {
    float inv = 1.0f / l_run[r];
    int t = b*S_ + q0 + wv*16 + r + 8*half;
    #pragma unroll
    for (int j = 0; j < 8; ++j)
      attn[(size_t)t*(H_*HD_) + head*HD_ + j*16 + l16] = (bf16)(oacc[j][r]*inv);
  }
}

extern "C" void kernel_launch(void* const* d_in, const int* in_sizes, int n_in,
                              void* d_out, int out_size, void* d_ws, size_t ws_size,
                              hipStream_t stream) {
  (void)in_sizes; (void)n_in; (void)out_size; (void)ws_size;
  const float* hs      = (const float*)d_in[0];
  const float* cosb    = (const float*)d_in[1];
  const float* sinb    = (const float*)d_in[2];
  const float* w_qkv   = (const float*)d_in[3];
  const float* b_qkv   = (const float*)d_in[4];
  const float* w_dense = (const float*)d_in[5];
  const float* b_dense = (const float*)d_in[6];

  char* ws = (char*)d_ws;
  bf16* hsb     = (bf16*)(ws);                    // 32 MB: T x D
  bf16* wqkvb   = (bf16*)(ws +  33554432ull);     // 48 MB: QKVN x D
  bf16* wdenseb = (bf16*)(ws +  83886080ull);     // 32 MB: D x D
  bf16* qkvraw  = (bf16*)(ws + 117440512ull);     // 48 MB: T x QKVN
  bf16* attnb   = (bf16*)(ws + 167772160ull);     // 32 MB: T x (H*HD)

  // 1) convert inputs to bf16
  cvt_f32_bf16<<<(T_*D_/4 + 255)/256, 256, 0, stream>>>(hs, hsb, T_*D_/4);
  cvt_f32_bf16<<<(QKVN*D_/4 + 255)/256, 256, 0, stream>>>(w_qkv, wqkvb, QKVN*D_/4);
  cvt_f32_bf16<<<(D_*D_/4 + 255)/256, 256, 0, stream>>>(w_dense, wdenseb, D_*D_/4);

  // 2) QKV projection: qkvraw = hs @ w_qkv^T + b_qkv (bf16 out)
  gemm_bf16<true><<<dim3(QKVN/128, T_/128), 256, 0, stream>>>(
      hsb, wqkvb, b_qkv, (void*)qkvraw, T_, QKVN, D_);

  // 3) in-place RoPE on q/k sections
  rope_kernel<<<(T_*40*32)/256, 256, 0, stream>>>(qkvraw, cosb, sinb);

  // 4) causal GQA flash attention
  flash_kernel<<<dim3(S_/128, B_, H_), 256, 0, stream>>>(qkvraw, attnb);

  // 5) out projection: out = attn @ w_dense^T + b_dense (f32 out)
  gemm_bf16<false><<<dim3(D_/128, T_/128), 256, 0, stream>>>(
      attnb, wdenseb, b_dense, d_out, T_, D_, H_*HD_);
}
